// Attention_2757369004137
// MI455X (gfx1250) — matrix operand order repaired and checked by reading, not compiled, not run on hardware
//
#include <hip/hip_runtime.h>

typedef __attribute__((ext_vector_type(16))) _Float16 v16h;
typedef __attribute__((ext_vector_type(8)))  float    v8f;
typedef __attribute__((ext_vector_type(4)))  unsigned int v4u;
typedef __attribute__((ext_vector_type(8)))  int      v8i;
typedef __attribute__((ext_vector_type(4)))  int      v4i;

#define BATCH 4
#define HEADS 16
#define SEQ   2048
#define DHEAD 64
#define NEGV  (-1000000000.0f)

#define SSTR 2052   // padded LDS stride (floats) for 16 x SEQ score rows
#define QSTR 68     // padded LDS stride for 16 x 64 Q tile (TDM pad 64+4 dwords)

__global__ __launch_bounds__(256) void attn_fused_kernel(
    const float* __restrict__ Q, const float* __restrict__ K,
    const float* __restrict__ V, const int*   __restrict__ M,
    float* __restrict__ Out, float* __restrict__ P)
{
  extern __shared__ char smem_raw[];
  float* sS = (float*)smem_raw;        // 16 x SSTR  (scores -> probabilities)
  float* sQ = sS + 16 * SSTR;          // 16 x QSTR  (Q tile, f32, TDM-padded)

  const int bh   = blockIdx.y;         // b*H + h
  const int b    = bh >> 4;            // H == 16
  const int q0   = blockIdx.x * 16;
  const int tid  = threadIdx.x;
  const int w    = tid >> 5;           // wave id 0..7 (wave32)
  const int lane = tid & 31;

  const float* Qb = Q + (size_t)bh * SEQ * DHEAD;
  const float* Kb = K + (size_t)bh * SEQ * DHEAD;
  const float* Vb = V + (size_t)bh * SEQ * DHEAD;
  const int*   Mb = M + (size_t)b  * SEQ * SEQ;

  // ---- stage Q tile (16 x 64 f32) into LDS via the Tensor Data Mover ----
  // 2D descriptor: tile 64x16 elements of 4B, row stride 64 elements in memory,
  // LDS pad of 4 dwords after every 64 dwords -> LDS row stride = QSTR = 68.
  if (w == 0) {
    const unsigned long long ga =
        (unsigned long long)(const void*)(Qb + (size_t)q0 * DHEAD);
    const unsigned lds_q = (unsigned)(16u * SSTR * 4u);  // byte offset of sQ
    v4u g0;
    g0[0] = 1u;                                  // count=1 (valid), user mode
    g0[1] = lds_q;                               // lds_addr[31:0]
    g0[2] = (unsigned)(ga & 0xffffffffu);        // global_addr[31:0]
    g0[3] = (unsigned)((ga >> 32) & 0x01ffffffu) // global_addr[56:32]
            | (2u << 30);                        // type = 2 ("image")
    v8i g1;
    g1[0] = (2 << 16)     // data_size = 2 -> 4 bytes
          | (1 << 20)     // pad_enable
          | (5 << 22)     // pad_interval: 64 dwords
          | (3 << 25);    // pad_amount: 4 dwords
    g1[1] = (DHEAD & 0xffff) << 16;  // tensor_dim0[15:0] = 64
    g1[2] = (16 << 16);              // tensor_dim0 hi = 0 | tensor_dim1 lo = 16
    g1[3] = (DHEAD << 16);           // tensor_dim1 hi = 0 | tile_dim0 = 64
    g1[4] = 16;                      // tile_dim1 = 16, tile_dim2 = 0
    g1[5] = DHEAD;                   // tensor_dim0_stride[31:0] = 64
    g1[6] = 0;                       // stride0 hi | tensor_dim1_stride lo
    g1[7] = 0;                       // tensor_dim1_stride hi
    const v4i gz4 = {0, 0, 0, 0};
    const v8i gz8 = {0, 0, 0, 0, 0, 0, 0, 0};
    __builtin_amdgcn_tensor_load_to_lds(g0, g1, gz4, gz4, gz8, 0);
    __builtin_amdgcn_s_wait_tensorcnt(0);
  }
  __syncthreads();

  const int  m16 = lane & 15;
  const bool hi  = (lane >= 16);
  const int  kbA = hi ? 8  : 0;   // A-fragment K sub-base (16-bit A layout)
  const int  kbB = hi ? 16 : 0;   // A/B-fragment K base (contiguous k = 16h + j)

  // ---- per-wave Q A-fragments (constant over all k-tiles) ----
  v16h qa_lo, qa_hi;
#pragma unroll
  for (int j = 0; j < 16; ++j) {
    const int kk = kbB + j;   // K in [0,32)
    qa_lo[j] = (_Float16)sQ[m16 * QSTR + kk];
    qa_hi[j] = (_Float16)sQ[m16 * QSTR + 32 + kk];
  }

  // ---- phase 1: scores S = QK^T / 8, masked, into LDS ----
  for (int kt = w; kt < SEQ / 16; kt += 8) {
    const int k0 = kt * 16;
    // B fragment: B[d][n] = K[k0+n][d]; this lane: n = m16, d in [kbB, kbB+16) (+32)
    const float4* kp4 = (const float4*)(Kb + (size_t)(k0 + m16) * DHEAD + kbB);
    v16h kb_lo, kb_hi;
#pragma unroll
    for (int j4 = 0; j4 < 4; ++j4) {
      const float4 lo = kp4[j4];       // d = kbB + 4*j4 ..
      const float4 h4 = kp4[j4 + 8];   // d = kbB + 32 + 4*j4 ..
      kb_lo[4 * j4 + 0] = (_Float16)lo.x; kb_lo[4 * j4 + 1] = (_Float16)lo.y;
      kb_lo[4 * j4 + 2] = (_Float16)lo.z; kb_lo[4 * j4 + 3] = (_Float16)lo.w;
      kb_hi[4 * j4 + 0] = (_Float16)h4.x; kb_hi[4 * j4 + 1] = (_Float16)h4.y;
      kb_hi[4 * j4 + 2] = (_Float16)h4.z; kb_hi[4 * j4 + 3] = (_Float16)h4.w;
    }

    if (kt + 8 < SEQ / 16)
      __builtin_prefetch(Kb + (size_t)(k0 + 128 + m16) * DHEAD + kbB, 0, 1);

    v8f c = {};
    c = __builtin_amdgcn_wmma_f32_16x16x32_f16(false, qa_lo, false, kb_lo,
                                               (short)0, c, false, false);
    c = __builtin_amdgcn_wmma_f32_16x16x32_f16(false, qa_hi, false, kb_hi,
                                               (short)0, c, false, false);

    // C layout: c[r] -> row (r or r+8), col = m16
#pragma unroll
    for (int r = 0; r < 8; ++r) {
      const int qr = r + (hi ? 8 : 0);
      const int kc = k0 + m16;
      float s = c[r] * 0.125f;                              // 1/sqrt(64)
      s = (Mb[(size_t)(q0 + qr) * SEQ + kc] == 0) ? NEGV : s;
      sS[qr * SSTR + kc] = s;
    }
  }
  __syncthreads();

  // ---- phase 2: row softmax in LDS (each wave owns 2 rows) ----
#pragma unroll
  for (int rr = 0; rr < 2; ++rr) {
    const int r = 2 * w + rr;
    float* row = sS + r * SSTR;
    float mx = -3.0e38f;
    for (int i = lane; i < SEQ; i += 32) mx = fmaxf(mx, row[i]);
#pragma unroll
    for (int off = 16; off > 0; off >>= 1) mx = fmaxf(mx, __shfl_xor(mx, off, 32));
    float sum = 0.0f;
    for (int i = lane; i < SEQ; i += 32) {
      const float e = exp2f((row[i] - mx) * 1.4426950408889634f);
      row[i] = e;
      sum += e;
    }
#pragma unroll
    for (int off = 16; off > 0; off >>= 1) sum += __shfl_xor(sum, off, 32);
    const float inv = 1.0f / sum;
    for (int i = lane; i < SEQ; i += 32) row[i] *= inv;
  }
  __syncthreads();

  // ---- phase 3: waves 0-3 do PV WMMAs; waves 4-7 stream p_attn to HBM ----
  if (w < 4) {
    const int nt = w;                      // 16-wide d-tile
    v8f acc = {};
    for (int ks = 0; ks < SEQ / 32; ++ks) {
      const int k0 = ks * 32;
      v16h pa, vbf;
#pragma unroll
      for (int j = 0; j < 16; ++j) {
        const int kk = kbB + j;
        pa[j]  = (_Float16)sS[m16 * SSTR + k0 + kk];
        vbf[j] = (_Float16)Vb[(size_t)(k0 + kbB + j) * DHEAD + nt * 16 + m16];
      }
      acc = __builtin_amdgcn_wmma_f32_16x16x32_f16(false, pa, false, vbf,
                                                   (short)0, acc, false, false);
    }
#pragma unroll
    for (int r = 0; r < 8; ++r) {
      const int qr = r + (hi ? 8 : 0);
      Out[(size_t)bh * SEQ * DHEAD + (size_t)(q0 + qr) * DHEAD + nt * 16 + m16] =
          acc[r];
    }
  } else {
    // Stream normalized P from LDS straight to HBM with the async engine:
    // GLOBAL_STORE_ASYNC_FROM_LDS_B128, tracked on ASYNCcnt (no VGPR data path).
    const int r0 = (w - 4) * 4;            // 4 rows per streaming wave
#pragma unroll
    for (int rr = 0; rr < 4; ++rr) {
      const int r = r0 + rr;
      float* dstrow = P + ((size_t)bh * SEQ + q0 + r) * SEQ;
      for (int i = lane; i < SEQ / 4; i += 32) {
        const unsigned lds_off =
            (unsigned)(((unsigned)r * SSTR + (unsigned)i * 4u) * 4u); // bytes, 16B-aligned
        const unsigned long long gaddr =
            (unsigned long long)(void*)(dstrow + (size_t)i * 4);
        asm volatile("global_store_async_from_lds_b128 %0, %1, off"
                     :: "v"(gaddr), "v"(lds_off) : "memory");
      }
    }
    asm volatile("s_wait_asynccnt 0" ::: "memory");
  }
}

extern "C" void kernel_launch(void* const* d_in, const int* in_sizes, int n_in,
                              void* d_out, int out_size, void* d_ws, size_t ws_size,
                              hipStream_t stream) {
  (void)in_sizes; (void)n_in; (void)out_size; (void)d_ws; (void)ws_size;
  const float* Q = (const float*)d_in[0];
  const float* K = (const float*)d_in[1];
  const float* V = (const float*)d_in[2];
  const int*   M = (const int*)d_in[3];
  float* Out = (float*)d_out;                                  // [B,H,S,D]
  float* P   = Out + (size_t)BATCH * HEADS * SEQ * DHEAD;      // [B,H,S,S]

  const size_t lds_bytes = (size_t)(16 * SSTR + 16 * QSTR) * sizeof(float); // ~136 KB
  dim3 grid(SEQ / 16, BATCH * HEADS);   // 128 x 64 = 8192 workgroups
  dim3 block(256);                      // 8 wave32 waves
  attn_fused_kernel<<<grid, block, lds_bytes, stream>>>(Q, K, V, M, Out, P);
}
